// GGNNv2_LCG_15839839387878
// MI455X (gfx1250) — matrix operand order repaired and checked by reading, not compiled, not run on hardware
//
#include <hip/hip_runtime.h>
#include <hip/hip_bf16.h>

typedef __attribute__((ext_vector_type(16))) _Float16 v16h;
typedef __attribute__((ext_vector_type(8)))  float    v8f;

#define DIM 128
#define NITER 3

// ---------------------------------------------------------------------------
// f32 -> f16 weight conversion (runs once per launch over ~2MB of weights)
// ---------------------------------------------------------------------------
__global__ void f32_to_f16_kernel(const float* __restrict__ src,
                                  _Float16* __restrict__ dst, int n) {
    int i = blockIdx.x * blockDim.x + threadIdx.x;
    if (i < n) dst[i] = (_Float16)src[i];
}

// ---------------------------------------------------------------------------
// Zero-fill (float4)
// ---------------------------------------------------------------------------
__global__ void zero_f32_kernel(float4* __restrict__ p, int n4) {
    int i = blockIdx.x * blockDim.x + threadIdx.x;
    if (i < n4) p[i] = make_float4(0.f, 0.f, 0.f, 0.f);
}

// ---------------------------------------------------------------------------
// WMMA GEMM:  Cout[rows, O] = act( A[rows, K] @ W[O, K]^T + bias )
//   - one wave per 64x64 output macro-tile: acc[4 row-tiles][4 col-tiles]
//   - each 32-wide K chunk: load 4 B fragments ONCE (reused by 4 row tiles),
//     then per row-tile load+convert one A fragment and issue 4 WMMAs
//     -> 24 b128 loads : 16 v_wmma per chunk, f32->f16 cvt co-executes
//        with the matrix pipe (16-bit WMMA is TRANS-class on CDNA5)
//   - A f32 converted inline in contiguous K order (same as B):
//       lanes 0-15 : K-offsets {0..15}; lanes 16-31: {16..31}
//   - B (= W rows, f16) contiguous: lanes 0-15 -> K 0..15, 16-31 -> K 16..31
//   - concat-A mode (ksplit): k < ksplit reads A0 else A1
//   - swapA: read row (m^1) (literal <-> negated-literal pairing)
// ---------------------------------------------------------------------------
__global__ __launch_bounds__(32)
void wmma_gemm_kernel(const float* __restrict__ A0,
                      const float* __restrict__ A1,
                      int ksplit, int lda0, int lda1,
                      const _Float16* __restrict__ W,
                      const float* __restrict__ bias,
                      float* __restrict__ Cout,
                      int rows, int K, int O, int relu, int swapA) {
    const int lane = threadIdx.x;
    const int l15  = lane & 15;
    const int lh   = lane >> 4;           // 0 or 1
    const int m0   = blockIdx.x * 64;
    const int n0   = blockIdx.y * 64;

    // per-row-tile A row index (swap, then clamp for the ragged last block)
    int mrow[4];
#pragma unroll
    for (int mt = 0; mt < 4; ++mt) {
        int mm = m0 + mt * 16 + l15;
        if (swapA) mm ^= 1;
        if (mm >= rows) mm = rows - 1;
        mrow[mt] = mm;
    }

    v8f acc[4][4];
#pragma unroll
    for (int mt = 0; mt < 4; ++mt)
#pragma unroll
        for (int t = 0; t < 4; ++t) acc[mt][t] = v8f{};

    for (int k0 = 0; k0 < K; k0 += 32) {
        // ---- B fragments: loaded once, reused by all 4 row tiles ----
        v16h b[4];
#pragma unroll
        for (int t = 0; t < 4; ++t) {
            const _Float16* pb = W + (size_t)(n0 + t * 16 + l15) * K + k0 + lh * 16;
            b[t] = *(const v16h*)pb;      // 32B aligned (K mult of 32)
        }

        // ---- A-pointer select for concat mode ----
        const float* Abase;
        int ka, lda;
        if (k0 < ksplit) { Abase = A0; ka = k0;          lda = lda0; }
        else             { Abase = A1; ka = k0 - ksplit; lda = lda1; }

#pragma unroll
        for (int mt = 0; mt < 4; ++mt) {
            const float* pa = Abase + (size_t)mrow[mt] * lda + ka + lh * 16;
            v16h a;
#pragma unroll
            for (int j = 0; j < 8; ++j) {
                a[j]     = (_Float16)pa[j];
                a[j + 8] = (_Float16)pa[j + 8];
            }
#pragma unroll
            for (int t = 0; t < 4; ++t) {
                acc[mt][t] = __builtin_amdgcn_wmma_f32_16x16x32_f16(
                    false, a, false, b[t], (short)0, acc[mt][t], false, false);
            }
        }
    }

    // Epilogue: C layout -> VGPR r: lanes 0-15 = (M=+r, N=n), lanes 16-31 = (M=+8+r)
#pragma unroll
    for (int t = 0; t < 4; ++t) {
        const int n = n0 + t * 16 + l15;
        const float bv = bias[n];
#pragma unroll
        for (int mt = 0; mt < 4; ++mt) {
#pragma unroll
            for (int r = 0; r < 8; ++r) {
                const int row = m0 + mt * 16 + r + lh * 8;
                if (row < rows) {
                    float v = acc[mt][t][r] + bv;
                    if (relu) v = fmaxf(v, 0.0f);
                    Cout[(size_t)row * O + n] = v;
                }
            }
        }
    }
}

// ---------------------------------------------------------------------------
// Segment-sum: dst[didx[e], :] += src[sidx[e], :]  (one thread per edge*float4)
// One wave handles one edge -> index loads coalesce to a single line request.
// Scatter targets (51-102MB) fit in the 192MB L2 -> atomics resolve in L2.
// ---------------------------------------------------------------------------
__global__ void scatter_add_kernel(const float* __restrict__ src,
                                   const int* __restrict__ sidx,
                                   const int* __restrict__ didx,
                                   float* __restrict__ dst, int nEdges) {
    long long gid = (long long)blockIdx.x * blockDim.x + threadIdx.x;
    long long total = (long long)nEdges * 32;   // 128 floats / 4 per thread
    if (gid >= total) return;
    int e = (int)(gid >> 5);
    int g = (int)(gid & 31);
    const float4 v = *(const float4*)(src + (size_t)sidx[e] * DIM + g * 4);
    float* dp = dst + (size_t)didx[e] * DIM + g * 4;
    unsafeAtomicAdd(dp + 0, v.x);
    unsafeAtomicAdd(dp + 1, v.y);
    unsafeAtomicAdd(dp + 2, v.z);
    unsafeAtomicAdd(dp + 3, v.w);
}

// ---------------------------------------------------------------------------
// GRU gates (float4): h' = (1-z)*n + z*h  (gate order r,z,n; gi/gh [rows,384])
// ---------------------------------------------------------------------------
__global__ void gru_gate_kernel(const float4* __restrict__ gi,
                                const float4* __restrict__ gh,
                                const float4* __restrict__ h,
                                float4* __restrict__ out, int n4) {
    int idx = blockIdx.x * blockDim.x + threadIdx.x;
    if (idx >= n4) return;
    int row = idx >> 5;                   // 32 float4 per 128-wide row
    int d4  = idx & 31;
    size_t b = (size_t)row * 96 + d4;     // 96 float4 per 384-wide row
    float4 ir = gi[b], iz = gi[b + 32], inn = gi[b + 64];
    float4 hr = gh[b], hz = gh[b + 32], hn  = gh[b + 64];
    float4 hv = h[idx];
    float4 o;
    {
        float r = 1.0f / (1.0f + __expf(-(ir.x + hr.x)));
        float z = 1.0f / (1.0f + __expf(-(iz.x + hz.x)));
        float ng = tanhf(inn.x + r * hn.x);
        o.x = (1.0f - z) * ng + z * hv.x;
    }
    {
        float r = 1.0f / (1.0f + __expf(-(ir.y + hr.y)));
        float z = 1.0f / (1.0f + __expf(-(iz.y + hz.y)));
        float ng = tanhf(inn.y + r * hn.y);
        o.y = (1.0f - z) * ng + z * hv.y;
    }
    {
        float r = 1.0f / (1.0f + __expf(-(ir.z + hr.z)));
        float z = 1.0f / (1.0f + __expf(-(iz.z + hz.z)));
        float ng = tanhf(inn.z + r * hn.z);
        o.z = (1.0f - z) * ng + z * hv.z;
    }
    {
        float r = 1.0f / (1.0f + __expf(-(ir.w + hr.w)));
        float z = 1.0f / (1.0f + __expf(-(iz.w + hz.w)));
        float ng = tanhf(inn.w + r * hn.w);
        o.w = (1.0f - z) * ng + z * hv.w;
    }
    out[idx] = o;
}

// ---------------------------------------------------------------------------
// Host-side orchestration
// ---------------------------------------------------------------------------
extern "C" void kernel_launch(void* const* d_in, const int* in_sizes, int n_in,
                              void* d_out, int out_size, void* d_ws, size_t ws_size,
                              hipStream_t stream) {
    (void)n_in; (void)out_size; (void)ws_size;

    const int* l_edge = (const int*)d_in[2];
    const int* c_edge = (const int*)d_in[3];
    const float* l_emb0 = (const float*)d_in[4];
    const float* c_emb0 = (const float*)d_in[5];
    const int L = in_sizes[4] / DIM;
    const int C = in_sizes[5] / DIM;
    const int E = in_sizes[2];

    // params in setup_inputs() dict order
    enum { L2C_W1, L2C_B1, L2C_W2, L2C_B2,
           C2L_W1, C2L_B1, C2L_W2, C2L_B2,
           L2L_W1, L2L_B1, L2L_W2, L2L_B2,
           CG_WIH, CG_WHH, CG_BIH, CG_BHH,
           LG_WIH, LG_WHH, LG_BIH, LG_BHH, NPARAM };
    const float* P[NPARAM];
    for (int i = 0; i < NPARAM; ++i) P[i] = (const float*)d_in[6 + i];

    const size_t fL = (size_t)L * DIM;
    const size_t fC = (size_t)C * DIM;
    const size_t mxR = (size_t)(L > C ? L : C);

    // ---- workspace carve-up (f32 region, then f16 weight region) ----
    float* wsf  = (float*)d_ws;
    float* hid  = wsf;  wsf += mxR * DIM;      // MLP hidden (max rows)
    float* lmsg = wsf;  wsf += fL;             // l2c message features
    float* cmsg = wsf;  wsf += fC;             // c2l message features
    float* l2lm = wsf;  wsf += fL;             // l2l messages
    float* aggC = wsf;  wsf += fC;             // segment-sum into clauses
    float* aggL = wsf;  wsf += fL;             // segment-sum into literals
    float* gi   = wsf;  wsf += mxR * 3 * DIM;  // GRU input gates
    float* gh   = wsf;  wsf += mxR * 3 * DIM;  // GRU hidden gates

    _Float16* hp = (_Float16*)wsf;             // 512B-aligned (offsets mult of 128 f32)
    const size_t szMLP = (size_t)NITER * DIM * DIM;
    const size_t szW3  = (size_t)NITER * 3 * DIM * DIM;
    const size_t szW3b = (size_t)NITER * 3 * DIM * 2 * DIM;
    _Float16* w_l2c1 = hp; hp += szMLP;
    _Float16* w_l2c2 = hp; hp += szMLP;
    _Float16* w_c2l1 = hp; hp += szMLP;
    _Float16* w_c2l2 = hp; hp += szMLP;
    _Float16* w_l2l1 = hp; hp += szMLP;
    _Float16* w_l2l2 = hp; hp += szMLP;
    _Float16* w_cwih = hp; hp += szW3;
    _Float16* w_cwhh = hp; hp += szW3;
    _Float16* w_lwih = hp; hp += szW3b;
    _Float16* w_lwhh = hp; hp += szW3;

    auto cvt = [&](const float* s, _Float16* d, size_t n) {
        f32_to_f16_kernel<<<(unsigned)((n + 255) / 256), 256, 0, stream>>>(s, d, (int)n);
    };
    cvt(P[L2C_W1], w_l2c1, szMLP);  cvt(P[L2C_W2], w_l2c2, szMLP);
    cvt(P[C2L_W1], w_c2l1, szMLP);  cvt(P[C2L_W2], w_c2l2, szMLP);
    cvt(P[L2L_W1], w_l2l1, szMLP);  cvt(P[L2L_W2], w_l2l2, szMLP);
    cvt(P[CG_WIH], w_cwih, szW3);   cvt(P[CG_WHH], w_cwhh, szW3);
    cvt(P[LG_WIH], w_lwih, szW3b);  cvt(P[LG_WHH], w_lwhh, szW3);

    // ---- output: stacked l_embs [(T+1),L,DIM] then c_embs [(T+1),C,DIM] ----
    float* outL = (float*)d_out;
    float* outC = outL + (size_t)(NITER + 1) * fL;
    hipMemcpyAsync(outL, l_emb0, fL * sizeof(float), hipMemcpyDeviceToDevice, stream);
    hipMemcpyAsync(outC, c_emb0, fC * sizeof(float), hipMemcpyDeviceToDevice, stream);

    const float* cur_l = outL;
    const float* cur_c = outC;

    auto gemm = [&](const float* A0, const float* A1, int ksplit, int lda0, int lda1,
                    const _Float16* W, const float* b, float* Co,
                    int rows, int K, int O, int relu, int swap) {
        dim3 grid((unsigned)((rows + 63) / 64), (unsigned)(O / 64));
        wmma_gemm_kernel<<<grid, dim3(32), 0, stream>>>(
            A0, A1, ksplit, lda0, lda1, W, b, Co, rows, K, O, relu, swap);
    };
    auto zero = [&](float* p, size_t n) {
        zero_f32_kernel<<<(unsigned)((n / 4 + 255) / 256), 256, 0, stream>>>(
            (float4*)p, (int)(n / 4));
    };
    auto scatter = [&](const float* src, const int* si, const int* di, float* dst) {
        long long tot = (long long)E * 32;
        scatter_add_kernel<<<(unsigned)((tot + 255) / 256), 256, 0, stream>>>(src, si, di, dst, E);
    };
    auto gate = [&](const float* gi_, const float* gh_, const float* h_, float* o_, size_t n) {
        int n4 = (int)(n / 4);
        gru_gate_kernel<<<(unsigned)((n4 + 255) / 256), 256, 0, stream>>>(
            (const float4*)gi_, (const float4*)gh_, (const float4*)h_, (float4*)o_, n4);
    };

    for (int i = 0; i < NITER; ++i) {
        const size_t wM = (size_t)i * DIM * DIM;
        const size_t bD = (size_t)i * DIM;
        const size_t w3 = (size_t)i * 3 * DIM * DIM;
        const size_t b3 = (size_t)i * 3 * DIM;
        const size_t w3b = (size_t)i * 3 * DIM * 2 * DIM;

        // literal -> clause message MLP
        gemm(cur_l, nullptr, DIM, DIM, DIM, w_l2c1 + wM, P[L2C_B1] + bD, hid, L, DIM, DIM, 1, 0);
        gemm(hid,   nullptr, DIM, DIM, DIM, w_l2c2 + wM, P[L2C_B2] + bD, lmsg, L, DIM, DIM, 0, 0);
        // clause -> literal message MLP
        gemm(cur_c, nullptr, DIM, DIM, DIM, w_c2l1 + wM, P[C2L_B1] + bD, hid, C, DIM, DIM, 1, 0);
        gemm(hid,   nullptr, DIM, DIM, DIM, w_c2l2 + wM, P[C2L_B2] + bD, cmsg, C, DIM, DIM, 0, 0);
        // literal <-> negated-literal MLP (row swap fused into A load)
        gemm(cur_l, nullptr, DIM, DIM, DIM, w_l2l1 + wM, P[L2L_B1] + bD, hid, L, DIM, DIM, 1, 1);
        gemm(hid,   nullptr, DIM, DIM, DIM, w_l2l2 + wM, P[L2L_B2] + bD, l2lm, L, DIM, DIM, 0, 0);

        // aggregate literal messages into clauses, clause GRU
        zero(aggC, fC);
        scatter(lmsg, l_edge, c_edge, aggC);
        gemm(aggC,  nullptr, DIM, DIM, DIM, w_cwih + w3, P[CG_BIH] + b3, gi, C, DIM, 3 * DIM, 0, 0);
        gemm(cur_c, nullptr, DIM, DIM, DIM, w_cwhh + w3, P[CG_BHH] + b3, gh, C, DIM, 3 * DIM, 0, 0);
        float* next_c = outC + (size_t)(i + 1) * fC;
        gate(gi, gh, cur_c, next_c, fC);
        cur_c = next_c;

        // aggregate clause messages into literals, literal GRU (K=256 concat-A)
        zero(aggL, fL);
        scatter(cmsg, c_edge, l_edge, aggL);
        gemm(aggL, l2lm, DIM, DIM, DIM, w_lwih + w3b, P[LG_BIH] + b3, gi, L, 2 * DIM, 3 * DIM, 0, 0);
        gemm(cur_l, nullptr, DIM, DIM, DIM, w_lwhh + w3, P[LG_BHH] + b3, gh, L, DIM, 3 * DIM, 0, 0);
        float* next_l = outL + (size_t)(i + 1) * fL;
        gate(gi, gh, cur_l, next_l, fL);
        cur_l = next_l;
    }
}
